// Cross_Attention_49246095016452
// MI455X (gfx1250) — matrix operand order repaired and checked by reading, not compiled, not run on hardware
//
#include <hip/hip_runtime.h>

typedef _Float16 v16h __attribute__((ext_vector_type(16)));
typedef float    v8f  __attribute__((ext_vector_type(8)));

#define N_TOK 4096
#define SEQ   4097
#define SC_F  0.125f                  // DH^-0.5, DH=64
#define SQD_F 8.0f                    // sqrt(64)
#define SQM_F 1.7320508075688772f     // sqrt(3)

// ---------------- block reduction helpers (blockDim == 256, 8 waves) -------
__device__ __forceinline__ float warpSum(float v) {
#pragma unroll
  for (int off = 16; off > 0; off >>= 1) v += __shfl_down(v, off, 32);
  return v;
}
__device__ __forceinline__ float warpMax(float v) {
#pragma unroll
  for (int off = 16; off > 0; off >>= 1) v = fmaxf(v, __shfl_down(v, off, 32));
  return v;
}
__device__ __forceinline__ float blockSum(float v, float* s8) {
  const int lane = threadIdx.x & 31, w = threadIdx.x >> 5;
  v = warpSum(v);
  __syncthreads();
  if (lane == 0) s8[w] = v;
  __syncthreads();
  float r = 0.f;
#pragma unroll
  for (int i = 0; i < 8; ++i) r += s8[i];
  return r;
}
__device__ __forceinline__ float blockMax(float v, float* s8) {
  const int lane = threadIdx.x & 31, w = threadIdx.x >> 5;
  v = warpMax(v);
  __syncthreads();
  if (lane == 0) s8[w] = v;
  __syncthreads();
  float r = -3.4e38f;
#pragma unroll
  for (int i = 0; i < 8; ++i) r = fmaxf(r, s8[i]);
  return r;
}

// ---------------- prep: CLS proj + LN + q + sk/sv + u(f16) + small scores --
__global__ void prep_kernel(
    const float* __restrict__ cls, int din, int dout,
    const float* __restrict__ piW, const float* __restrict__ pib,
    const float* __restrict__ lng, const float* __restrict__ lnb,
    const float* __restrict__ Wq,  const float* __restrict__ bq,
    const float* __restrict__ Wkv, const float* __restrict__ bkv,
    const float* __restrict__ mk,  const float* __restrict__ Smk,
    _Float16* __restrict__ u16, float* __restrict__ svv,
    float* __restrict__ beta, float* __restrict__ scls,
    float* __restrict__ cmem, float* __restrict__ selfa)
{
  __shared__ float shx[1024];
  __shared__ float shq[512];
  __shared__ float shsk[512];
  __shared__ float s8[8];
  const int b = blockIdx.x, tid = threadIdx.x;
  const float* crow = cls + (size_t)b * din;

  // t = cls @ piW + pib
  for (int j = tid; j < dout; j += 256) {
    float acc = pib[j];
    for (int i = 0; i < din; ++i) acc += crow[i] * piW[(size_t)i * dout + j];
    shx[j] = acc;
  }
  __syncthreads();
  // LayerNorm over dout
  float s = 0.f, s2 = 0.f;
  for (int j = tid; j < dout; j += 256) { float v = shx[j]; s += v; s2 += v * v; }
  const float S1 = blockSum(s, s8);
  const float S2 = blockSum(s2, s8);
  const float mu = S1 / (float)dout;
  const float inv = rsqrtf(S2 / (float)dout - mu * mu + 1e-5f);
  __syncthreads();
  for (int j = tid; j < dout; j += 256)
    shx[j] = (shx[j] - mu) * inv * lng[j] + lnb[j];
  __syncthreads();
  // q = x @ Wq + bq   (INNER = 512)
  for (int c = tid; c < 512; c += 256) {
    float acc = bq[c];
    for (int i = 0; i < dout; ++i) acc += shx[i] * Wq[(size_t)i * 512 + c];
    shq[c] = acc;
  }
  // skv = x @ Wkv + bkv  (first 512 = sk, second 512 = sv)
  for (int c = tid; c < 1024; c += 256) {
    float acc = bkv[c];
    for (int i = 0; i < dout; ++i) acc += shx[i] * Wkv[(size_t)i * 1024 + c];
    if (c < 512) shsk[c] = acc;
    else         svv[(size_t)b * 512 + (c - 512)] = acc;
  }
  __syncthreads();
  // u[h][i] = sum_dd Wkv[i][h*64+dd] * q[h*64+dd]   (store f16, 16 padded cols)
  for (int i = tid; i < dout; i += 256) {
    const float* wr = Wkv + (size_t)i * 1024;
    float acc[8];
#pragma unroll
    for (int h = 0; h < 8; ++h) acc[h] = 0.f;
    for (int dd = 0; dd < 64; ++dd) {
#pragma unroll
      for (int h = 0; h < 8; ++h) acc[h] += wr[h * 64 + dd] * shq[h * 64 + dd];
    }
#pragma unroll
    for (int h = 0; h < 8; ++h)
      u16[((size_t)b * 16 + h) * dout + i] = (_Float16)acc[h];
  }
  // zero the padded columns 8..15
  for (int i = tid; i < 8 * dout; i += 256)
    u16[((size_t)b * 16 + 8) * dout + i] = (_Float16)0.f;
  // per-head scalars
  if (tid < 8) {
    const int h = tid;
    float bt = 0.f, sc = 0.f;
    for (int dd = 0; dd < 64; ++dd) {
      const float qv = shq[h * 64 + dd];
      bt += bkv[h * 64 + dd] * qv;
      sc += shsk[h * 64 + dd] * qv;
    }
    beta[b * 8 + h] = bt;
    scls[b * 8 + h] = sc;
    float sm[3];
    for (int m = 0; m < 3; ++m) {
      float a = 0.f, a2 = 0.f;
      for (int dd = 0; dd < 64; ++dd) {
        const float qv = shq[h * 64 + dd];
        a  += mk [h * 192 + m * 64 + dd] * qv;
        a2 += Smk[h * 192 + m * 64 + dd] * qv;
      }
      cmem[(b * 8 + h) * 3 + m] = a * SQD_F;
      sm[m] = a2 * SQD_F;
    }
    const float t0 = sc * SC_F, t1 = sm[0] * SC_F, t2 = sm[1] * SC_F, t3 = sm[2] * SC_F;
    const float mx = fmaxf(fmaxf(t0, t1), fmaxf(t2, t3));
    const float e0 = expf(t0 - mx), e1 = expf(t1 - mx), e2 = expf(t2 - mx), e3 = expf(t3 - mx);
    const float Z = e0 + e1 + e2 + e3;
    float* sa = selfa + (b * 8 + h) * 4;
    sa[0] = e0 / Z; sa[1] = e1 / Z; sa[2] = e2 / Z; sa[3] = e3 / Z;
  }
}

// ---------------- score: S[b,h,j] = patch[b,j,:] . u[b,h,:]  via WMMA ------
// One wave per 16-row tile; K looped in chunks of 32 (v_wmma_f32_16x16x32_f16).
__global__ void score_kernel(
    const float* __restrict__ tokens, int D,
    const _Float16* __restrict__ u16,
    float* __restrict__ S)
{
  const int b    = blockIdx.y;
  const int lane = threadIdx.x & 31;
  const int wave = threadIdx.x >> 5;
  const int tile = (blockIdx.x * 8 + wave) * 16;
  const int rlo  = lane & 15;           // A: row in tile / B: column (head)
  const int hi   = lane >> 4;           // lane half
  const int base = hi * 16;             // A-fragment K group
  const int kb   = hi * 16;             // B-fragment K group
  const float*     prow = tokens + ((size_t)(b * SEQ) + 1 + tile + rlo) * D;
  const _Float16*  ub   = u16 + ((size_t)b * 16 + rlo) * D;  // cols 8..15 are zeros

  v8f acc = {};
  for (int k0 = 0; k0 < D; k0 += 32) {
    const float4 x0 = *(const float4*)(prow + k0 + base);
    const float4 x1 = *(const float4*)(prow + k0 + base + 4);
    const float4 x2 = *(const float4*)(prow + k0 + base + 8);
    const float4 x3 = *(const float4*)(prow + k0 + base + 12);
    v16h a;
    a[0] = (_Float16)x0.x; a[1] = (_Float16)x0.y; a[2]  = (_Float16)x0.z; a[3]  = (_Float16)x0.w;
    a[4] = (_Float16)x1.x; a[5] = (_Float16)x1.y; a[6]  = (_Float16)x1.z; a[7]  = (_Float16)x1.w;
    a[8] = (_Float16)x2.x; a[9] = (_Float16)x2.y; a[10] = (_Float16)x2.z; a[11] = (_Float16)x2.w;
    a[12] = (_Float16)x3.x; a[13] = (_Float16)x3.y; a[14] = (_Float16)x3.z; a[15] = (_Float16)x3.w;
    const v16h bf = *(const v16h*)(ub + k0 + kb);
    acc = __builtin_amdgcn_wmma_f32_16x16x32_f16(false, a, false, bf,
                                                 (short)0, acc, false, false);
  }
  if (rlo < 8) {  // heads only
    float* dst = S + ((size_t)(b * 8) + rlo) * N_TOK + tile + hi * 8;
    *(float4*)(dst)     = make_float4(acc[0], acc[1], acc[2], acc[3]);
    *(float4*)(dst + 4) = make_float4(acc[4], acc[5], acc[6], acc[7]);
  }
}

// ---------------- reduce: softmax + fW weighting + output path + CLS update
__global__ void reduce_kernel(
    const float* __restrict__ S,
    const float* __restrict__ beta, const float* __restrict__ scls,
    const float* __restrict__ cmem, const float* __restrict__ selfa,
    const float* __restrict__ svv,  const float* __restrict__ Smv,
    const float* __restrict__ fW,   const float* __restrict__ fb,
    const float* __restrict__ Wo,   const float* __restrict__ bo,
    const float* __restrict__ poW,  const float* __restrict__ pob,
    float* __restrict__ cls, int din, int dout)
{
  __shared__ float shout[512];
  __shared__ float sho[1024];
  __shared__ float s8[8];
  const int b = blockIdx.x, tid = threadIdx.x;

  for (int h = 0; h < 8; ++h) {
    const float* Sr = S + ((size_t)(b * 8) + h) * N_TOK;
    const float bt = beta[b * 8 + h];
    float mx = -3.4e38f;
    for (int j = tid; j < N_TOK; j += 256) mx = fmaxf(mx, (Sr[j] + bt) * SC_F);
    mx = blockMax(mx, s8);
    const float sC  = scls[b * 8 + h] * SC_F;
    const float cm0 = cmem[(b * 8 + h) * 3 + 0] * SC_F;
    const float cm1 = cmem[(b * 8 + h) * 3 + 1] * SC_F;
    const float cm2 = cmem[(b * 8 + h) * 3 + 2] * SC_F;
    mx = fmaxf(mx, fmaxf(fmaxf(sC, cm0), fmaxf(cm1, cm2)));

    float z = 0.f, w0 = 0.f, w1 = 0.f, w2 = 0.f, w3 = 0.f;
    for (int j = tid; j < N_TOK; j += 256) {
      const float e = expf((Sr[j] + bt) * SC_F - mx);
      const float4 f = *(const float4*)(fW + (size_t)j * 4);
      z += e; w0 += e * f.x; w1 += e * f.y; w2 += e * f.z; w3 += e * f.w;
    }
    z  = blockSum(z,  s8);
    w0 = blockSum(w0, s8);
    w1 = blockSum(w1, s8);
    w2 = blockSum(w2, s8);
    w3 = blockSum(w3, s8);
    // extra rows: CLS (fW row 4096) and 3 memory rows (4097..4099)
    {
      const float e = expf(sC - mx);
      const float* f = fW + (size_t)4096 * 4;
      z += e; w0 += e * f[0]; w1 += e * f[1]; w2 += e * f[2]; w3 += e * f[3];
      const float cms[3] = { cm0, cm1, cm2 };
      for (int m = 0; m < 3; ++m) {
        const float em = expf(cms[m] - mx);
        const float* fm = fW + (size_t)(4097 + m) * 4;
        z += em; w0 += em * fm[0]; w1 += em * fm[1]; w2 += em * fm[2]; w3 += em * fm[3];
      }
    }
    const float* sa = selfa + (b * 8 + h) * 4;
    const float a0 = w0 / z + fb[0] + sa[0];
    const float a1 = w1 / z + fb[1] + sa[1];
    const float a2 = w2 / z + fb[2] + sa[2];
    const float a3 = w3 / z + fb[3] + sa[3];
    if (tid < 64) {
      const int dd = tid;
      shout[h * 64 + dd] =
          a0 * svv[(size_t)b * 512 + h * 64 + dd]
        + SQM_F * (a1 * Smv[h * 192 + dd]
                 + a2 * Smv[h * 192 + 64 + dd]
                 + a3 * Smv[h * 192 + 128 + dd]);
    }
  }
  __syncthreads();
  // o = out @ Wo + bo   (512 -> dout)
  for (int c = tid; c < dout; c += 256) {
    float acc = bo[c];
    for (int i = 0; i < 512; ++i) acc += shout[i] * Wo[(size_t)i * dout + c];
    sho[c] = acc;
  }
  __syncthreads();
  // cls += o @ poW + pob  (dout -> din)
  for (int c2 = tid; c2 < din; c2 += 256) {
    float acc = pob[c2];
    for (int i = 0; i < dout; ++i) acc += sho[i] * poW[(size_t)i * din + c2];
    cls[(size_t)b * din + c2] += acc;
  }
}

// ---------------- CLS copies ------------------------------------------------
__global__ void init_cls_kernel(const float* __restrict__ ht,
                                const float* __restrict__ lt,
                                float* __restrict__ hc, float* __restrict__ lc)
{
  const int t = blockIdx.x * 256 + threadIdx.x;
  if (t < 8 * 1024) { const int b = t >> 10, i = t & 1023; hc[t] = ht[(size_t)b * SEQ * 1024 + i]; }
  if (t < 8 * 512)  { const int b = t >> 9,  i = t & 511;  lc[t] = lt[(size_t)b * SEQ * 512  + i]; }
}
__global__ void write_cls_kernel(const float* __restrict__ hc,
                                 const float* __restrict__ lc,
                                 float* __restrict__ out)
{
  const int t = blockIdx.x * 256 + threadIdx.x;
  if (t < 8 * 1024) { const int b = t >> 10, i = t & 1023; out[(size_t)b * SEQ * 1024 + i] = hc[t]; }
  if (t < 8 * 512)  { const int b = t >> 9,  i = t & 511;
    out[(size_t)8 * SEQ * 1024 + (size_t)b * SEQ * 512 + i] = lc[t]; }
}

// ---------------- host orchestration ---------------------------------------
extern "C" void kernel_launch(void* const* d_in, const int* in_sizes, int n_in,
                              void* d_out, int out_size, void* d_ws, size_t ws_size,
                              hipStream_t stream)
{
  (void)in_sizes; (void)out_size; (void)ws_size;
  const float* ht = (const float*)d_in[0];
  const float* lt = (const float*)d_in[1];
  const float* P[38];
  for (int i = 0; i < 38 && i < n_in; ++i) P[i] = (const float*)d_in[i];
  enum { piW = 0, pib, poW, pob, lng, lnb, Wq, bq, Wkv, bkv, fW, fb, Wo, bo, mk, mv, Smk, Smv };
  const float* const* A  = P + 2;
  const float* const* Bp = P + 20;

  uintptr_t w = (uintptr_t)d_ws;
  auto take = [&](size_t bytes) -> void* {
    void* p = (void*)w; w += (bytes + 255) & ~(size_t)255; return p;
  };
  float* hc     = (float*)take((size_t)8 * 1024 * 4);
  float* lc     = (float*)take((size_t)8 * 512 * 4);
  float* svA    = (float*)take((size_t)8 * 512 * 4);
  float* svB    = (float*)take((size_t)8 * 512 * 4);
  float* betaA  = (float*)take(64 * 4);
  float* betaB  = (float*)take(64 * 4);
  float* sclsA  = (float*)take(64 * 4);
  float* sclsB  = (float*)take(64 * 4);
  float* cmemA  = (float*)take(192 * 4);
  float* cmemB  = (float*)take(192 * 4);
  float* selfaA = (float*)take(256 * 4);
  float* selfaB = (float*)take(256 * 4);
  float* S_A    = (float*)take((size_t)8 * 8 * 4096 * 4);
  float* S_B    = (float*)take((size_t)8 * 8 * 4096 * 4);
  _Float16* uA  = (_Float16*)take((size_t)8 * 16 * 512 * 2);
  _Float16* uB  = (_Float16*)take((size_t)8 * 16 * 1024 * 2);

  // bulk-copy tokens into output (patch rows are passthrough; CLS rows fixed later)
  hipMemcpyAsync(d_out, ht, (size_t)8 * SEQ * 1024 * sizeof(float),
                 hipMemcpyDeviceToDevice, stream);
  hipMemcpyAsync((float*)d_out + (size_t)8 * SEQ * 1024, lt,
                 (size_t)8 * SEQ * 512 * sizeof(float),
                 hipMemcpyDeviceToDevice, stream);

  init_cls_kernel<<<32, 256, 0, stream>>>(ht, lt, hc, lc);

  for (int d = 0; d < 2; ++d) {
    // ---- Branch A: cls = h_cls (din=1024), dout=512, patch = l_tokens (D=512)
    prep_kernel<<<8, 256, 0, stream>>>(hc, 1024, 512,
        A[piW] + (size_t)d * 1024 * 512, A[pib] + d * 512,
        A[lng] + d * 512, A[lnb] + d * 512,
        A[Wq]  + (size_t)d * 512 * 512,  A[bq]  + d * 512,
        A[Wkv] + (size_t)d * 512 * 1024, A[bkv] + d * 1024,
        A[mk]  + d * 1536, A[Smk] + d * 1536,
        uA, svA, betaA, sclsA, cmemA, selfaA);
    score_kernel<<<dim3(32, 8), 256, 0, stream>>>(lt, 512, uA, S_A);
    reduce_kernel<<<8, 256, 0, stream>>>(S_A, betaA, sclsA, cmemA, selfaA, svA,
        A[Smv] + d * 1536, A[fW] + (size_t)d * 4100 * 4, A[fb] + d * 4,
        A[Wo]  + (size_t)d * 512 * 512,  A[bo]  + d * 512,
        A[poW] + (size_t)d * 512 * 1024, A[pob] + d * 1024,
        hc, 1024, 512);

    // ---- Branch B: cls = l_cls (din=512), dout=1024, patch = h_tokens (D=1024)
    prep_kernel<<<8, 256, 0, stream>>>(lc, 512, 1024,
        Bp[piW] + (size_t)d * 512 * 1024, Bp[pib] + d * 1024,
        Bp[lng] + d * 1024, Bp[lnb] + d * 1024,
        Bp[Wq]  + (size_t)d * 1024 * 512,  Bp[bq]  + d * 512,
        Bp[Wkv] + (size_t)d * 1024 * 1024, Bp[bkv] + d * 1024,
        Bp[mk]  + d * 1536, Bp[Smk] + d * 1536,
        uB, svB, betaB, sclsB, cmemB, selfaB);
    score_kernel<<<dim3(32, 8), 256, 0, stream>>>(ht, 1024, uB, S_B);
    reduce_kernel<<<8, 256, 0, stream>>>(S_B, betaB, sclsB, cmemB, selfaB, svB,
        Bp[Smv] + d * 1536, Bp[fW] + (size_t)d * 4100 * 4, Bp[fb] + d * 4,
        Bp[Wo]  + (size_t)d * 512 * 1024, Bp[bo]  + d * 1024,
        Bp[poW] + (size_t)d * 1024 * 512, Bp[pob] + d * 512,
        lc, 512, 1024);
  }

  write_cls_kernel<<<32, 256, 0, stream>>>(hc, lc, (float*)d_out);
}
